// AttnFeatTransKCat_54090818126595
// MI455X (gfx1250) — compile-verified, matrix operand order checked by reading, not run on hardware
//
#include <hip/hip_runtime.h>
#include <hip/hip_bf16.h>

// Problem constants (match reference)
#define BB  256   // batch
#define CC  4     // in channels
#define NN  64    // neighbors
#define FF  256   // features
#define OCC 8     // out channels
#define OFF 256   // out features
// GEMM: M=BB=256, N=OCC*OFF=2048, K=2*CC*FF=2048

typedef __attribute__((ext_vector_type(16))) __bf16 v16bf;
typedef __attribute__((ext_vector_type(8)))  float  v8f;

// ---------------------------------------------------------------------------
// Kernel A: attention scalars + softmax + weighted neighbor sum  v[b,c,f]
// One workgroup (256 thr = 8 waves, wave32) per batch element.
// ---------------------------------------------------------------------------
__global__ __launch_bounds__(256) void attn_weights_kernel(
    const float* __restrict__ x, const float* __restrict__ neighbor,
    const float* __restrict__ Wa1, const float* __restrict__ Wa2,
    float* __restrict__ nbw)
{
  __shared__ float wa2s[FF];
  __shared__ float red[256];
  __shared__ float s2s[NN];
  __shared__ float wsh[NN];
  __shared__ float s1sh;

  const int b = blockIdx.x;
  const int t = threadIdx.x;

  // column sums of attention weight matrices (C=4 rows)
  float wa1_t = Wa1[t] + Wa1[FF + t] + Wa1[2 * FF + t] + Wa1[3 * FF + t];
  float wa2_t = Wa2[t] + Wa2[FF + t] + Wa2[2 * FF + t] + Wa2[3 * FF + t];
  wa2s[t] = wa2_t;

  // s1[b] = sum_{c,f} x[b,c,f] * wa1[f]
  const float* xb = x + (size_t)b * CC * FF;
  float p = 0.f;
#pragma unroll
  for (int c = 0; c < CC; ++c) p += xb[c * FF + t] * wa1_t;
  red[t] = p;
  __syncthreads();
  for (int s = 128; s > 0; s >>= 1) {
    if (t < s) red[t] += red[t + s];
    __syncthreads();
  }
  if (t == 0) s1sh = red[0];

  // s2[b,n]: each of 8 waves handles 8 neighbors, wave32 shuffle reduction
  const int wave = t >> 5, lane = t & 31;
  const float* nbase = neighbor + (size_t)b * NN * CC * FF;
  for (int i = 0; i < NN / 8; ++i) {
    const int n = wave * (NN / 8) + i;
    const float* nb_n = nbase + (size_t)n * CC * FF;
    float acc = 0.f;
#pragma unroll
    for (int c = 0; c < CC; ++c)
#pragma unroll
      for (int j = 0; j < FF / 32; ++j) {
        const int f = lane + 32 * j;
        acc += nb_n[c * FF + f] * wa2s[f];
      }
#pragma unroll
    for (int off = 16; off > 0; off >>= 1) acc += __shfl_down(acc, off, 32);
    if (lane == 0) s2s[n] = acc;
  }
  __syncthreads();

  // softmax over n of s1*s2[n]
  const float s1 = s1sh;
  if (t < NN) red[t] = s1 * s2s[t];
  __syncthreads();
  for (int s = 32; s > 0; s >>= 1) {
    if (t < s) red[t] = fmaxf(red[t], red[t + s]);
    __syncthreads();
  }
  const float m = red[0];
  __syncthreads();
  if (t < NN) { const float e = __expf(s1 * s2s[t] - m); wsh[t] = e; red[t] = e; }
  __syncthreads();
  for (int s = 32; s > 0; s >>= 1) {
    if (t < s) red[t] += red[t + s];
    __syncthreads();
  }
  const float inv = 1.f / red[0];
  __syncthreads();
  if (t < NN) wsh[t] *= inv;
  __syncthreads();

  // v[b,c,f] = sum_n w[n] * neighbor[b,n,c,f]  (second streaming pass; L2-hot)
  float acc0 = 0.f, acc1 = 0.f, acc2 = 0.f, acc3 = 0.f;
  for (int n = 0; n < NN; ++n) {
    const float wn = wsh[n];
    const float* nb_n = nbase + (size_t)n * CC * FF + t;
    if (n + 1 < NN) __builtin_prefetch(nb_n + CC * FF, 0, 0);  // global_prefetch_b8
    acc0 += wn * nb_n[0 * FF];
    acc1 += wn * nb_n[1 * FF];
    acc2 += wn * nb_n[2 * FF];
    acc3 += wn * nb_n[3 * FF];
  }
  float* o = nbw + (size_t)b * CC * FF + t;
  o[0 * FF] = acc0; o[1 * FF] = acc1; o[2 * FF] = acc2; o[3 * FF] = acc3;
}

// ---------------------------------------------------------------------------
// Kernel B: rank-1 adjacency.  One WG per (b,c).  fadj = u (x) v + v (x) u,
// sgnroot, column-normalize, write adj (268 MB — the roofline term), and
// accumulate h[t] = sum_x g(x,t) * rn[x]*u[x] using symmetry of g (no
// cross-lane reductions, fully coalesced stores).
// ---------------------------------------------------------------------------
__global__ __launch_bounds__(256) void adj_h_kernel(
    const float* __restrict__ x, const float* __restrict__ nbw,
    float* __restrict__ adjOut, float* __restrict__ hbuf)
{
  __shared__ float us[FF], vs[FF], qs[FF];
  const int bc = blockIdx.x;   // b*C + c
  const int t = threadIdx.x;   // owns column y = t

  us[t] = x[(size_t)bc * FF + t];
  vs[t] = nbw[(size_t)bc * FF + t];
  __syncthreads();
  const float ut = us[t], vt = vs[t];

  // pass 1: colsum[t] = sum_x |sgnroot(fadj[x,t])|
  float cs = 0.f;
  for (int xk = 0; xk < FF; ++xk) {
    const float tv = us[xk] * vt + ut * vs[xk];     // LDS broadcast reads
    float r = __fsqrt_rn(fmaxf(fabsf(tv), 1e-8f));
    if (tv == 0.f) r = 0.f;                         // jnp.sign(0)==0 semantics
    cs += r;
  }
  const float rn = 1.f / (cs + 1e-7f);
  qs[t] = rn * ut;
  __syncthreads();

  // pass 2: write adj + accumulate h
  float hacc = 0.f;
  float* arow = adjOut + (size_t)bc * FF * FF;
  for (int xk = 0; xk < FF; ++xk) {
    const float tv = us[xk] * vt + ut * vs[xk];
    float g = copysignf(__fsqrt_rn(fmaxf(fabsf(tv), 1e-8f)), tv);
    if (tv == 0.f) g = 0.f;
    float a = g * rn;
    if (a != a) a = 0.f;                            // NaN -> 0
    arow[(size_t)xk * FF + t] = a;                  // coalesced 1KB row stores
    hacc += g * qs[xk];                             // h[t] via symmetry of g
  }
  hbuf[(size_t)bc * FF + t] = hacc;
}

// ---------------------------------------------------------------------------
// Kernel C: out[b,o] = sum_k z[b,k] * Wc[o,k], z = [h ; x], via bf16 WMMA
// (v_wmma_f32_16x16x32_bf16, f32 accumulate). One wave per 16x16 output tile,
// K loop of 64 WMMAs. 2048 wave-tiles = 256 blocks x 8 waves.
// ---------------------------------------------------------------------------
__global__ __launch_bounds__(256) void out_gemm_kernel(
    const float* __restrict__ hbuf, const float* __restrict__ x,
    const float* __restrict__ Wc, float* __restrict__ out)
{
  const int lane = threadIdx.x & 31;
  const int gw   = (blockIdx.x * 256 + threadIdx.x) >> 5;   // global wave id
  const int NT   = (OCC * OFF) / 16;                        // 128 N-tiles
  const int m0   = (gw / NT) * 16;
  const int n0   = (gw % NT) * 16;
  const int K     = 2 * CC * FF;   // 2048
  const int halfK = CC * FF;       // 1024

  const int zrow = m0 + (lane & 15);
  const int wrow = n0 + (lane & 15);
  const int koff = (lane >> 4) * 16;   // half-wave covers second 16 of K-slab

  v8f acc = {};
  for (int k0 = 0; k0 < K; k0 += 32) {
    const float* zsrc = (k0 < halfK)
        ? (hbuf + (size_t)zrow * halfK + (k0 + koff))
        : (x    + (size_t)zrow * halfK + (k0 - halfK + koff));
    const float* wsrc = Wc + (size_t)wrow * K + (k0 + koff);
    v16bf a, bfr;
#pragma unroll
    for (int j = 0; j < 16; ++j) {
      a[j]   = (__bf16)zsrc[j];    // in-register f32 -> bf16
      bfr[j] = (__bf16)wsrc[j];
    }
    acc = __builtin_amdgcn_wmma_f32_16x16x32_bf16(
        /*neg_a=*/false, a, /*neg_b=*/false, bfr,
        /*c_mod=*/(short)0, acc, /*reuse_a=*/false, /*reuse_b=*/false);
  }

  // C/D layout: VGPR r -> (M = r | 8+r per half-wave, N = lane%16)
  const int nn    = n0 + (lane & 15);
  const int mbase = m0 + (lane >> 4) * 8;
#pragma unroll
  for (int r = 0; r < 8; ++r)
    out[(size_t)(mbase + r) * (OCC * OFF) + nn] = acc[r];
}

// ---------------------------------------------------------------------------
extern "C" void kernel_launch(void* const* d_in, const int* in_sizes, int n_in,
                              void* d_out, int out_size, void* d_ws, size_t ws_size,
                              hipStream_t stream) {
  const float* x   = (const float*)d_in[0];   // (B,C,F)
  const float* nbr = (const float*)d_in[1];   // (B,N,C,F)
  const float* Wa1 = (const float*)d_in[2];   // (C,F)
  const float* Wa2 = (const float*)d_in[3];   // (C,F)
  const float* Wc  = (const float*)d_in[4];   // (OC*OF, 2C, F)

  float* out = (float*)d_out;                       // (B,OC,OF) = 524288 f32
  float* adj = out + (size_t)BB * OCC * OFF;        // (B,C,F,F) = 67.1M f32

  float* nbw  = (float*)d_ws;                       // B*C*F f32 (1 MB)
  float* hbuf = nbw + (size_t)BB * CC * FF;         // B*C*F f32 (1 MB)

  attn_weights_kernel<<<BB, 256, 0, stream>>>(x, nbr, Wa1, Wa2, nbw);
  adj_h_kernel<<<BB * CC, 256, 0, stream>>>(x, nbw, adj, hbuf);
  out_gemm_kernel<<<(BB / 16) * ((OCC * OFF) / 16) / 8, 256, 0, stream>>>(hbuf, x, Wc, out);
}
